// TransformerBlock_20014547599574
// MI455X (gfx1250) — matrix operand order repaired and checked by reading, not compiled, not run on hardware
//
#include <hip/hip_runtime.h>
#include <hip/hip_bf16.h>
#include <math.h>

// ---------------------------------------------------------------------------
// Problem constants (from the reference): B=4, S=1024, H=16, D=64, M=1024, F=4096
// ---------------------------------------------------------------------------
#define SEQ   1024
#define NB    4
#define NH    16
#define HD    64
#define MDIM  1024
#define FDIM  4096
#define ROWS  4096        // B*S
#define LDQKV 3072        // 3*M

#define AS1 __attribute__((address_space(1)))
#define AS3 __attribute__((address_space(3)))

typedef __bf16 bf16;
typedef __attribute__((ext_vector_type(16))) __bf16 v16bf;
typedef __attribute__((ext_vector_type(8)))  float  v8f;
typedef __attribute__((ext_vector_type(4)))  unsigned int v4u;
typedef __attribute__((ext_vector_type(8)))  int    v8i;
typedef __attribute__((ext_vector_type(4)))  int    v4i;

#if __has_builtin(__builtin_amdgcn_global_load_async_to_lds_b128)
#define HAVE_ASYNC 1
#else
#define HAVE_ASYNC 0
#endif
#if __has_builtin(__builtin_amdgcn_tensor_load_to_lds)
#define HAVE_TDM 1
#else
#define HAVE_TDM 0
#endif

union Frag16 { uint4 u[2]; v16bf v; bf16 h[16]; };

__device__ __forceinline__ v8f wmma_bf16(v16bf a, v16bf b, v8f c) {
  // D(16x16,f32) = A(16x32,bf16) * B(32x16,bf16) + C
  return __builtin_amdgcn_wmma_f32_16x16x32_bf16(false, a, false, b,
                                                 (short)0, c, false, false);
}

__device__ __forceinline__ float gelu_tanh(float x) {
  float x3 = x * x * x;
  return 0.5f * x * (1.0f + tanhf(0.79788456080286536f * (x + 0.044715f * x3)));
}

__device__ __forceinline__ unsigned lds_offset_of(const void* p) {
  return (unsigned)(size_t)(AS3 const void*)p;
}

#if HAVE_ASYNC
// 16B async global->LDS copy (no VGPR round-trip; tracked by ASYNCcnt).
// Builtin expects (AS1 int4*, AS3 int4*, imm offset, imm cpol).
__device__ __forceinline__ void async_copy_b128(const void* g, void* l) {
  __builtin_amdgcn_global_load_async_to_lds_b128((AS1 v4i*)g, (AS3 v4i*)l, 0, 0);
}
#endif

#if HAVE_TDM
// One-shot 2D TDM DMA: rows x cols (bf16) tile, row stride in elements, into LDS.
// D# packing per CDNA5 ISA 8.3/8.4 (group0: count|lds|gaddr|type=2; group1: dims).
__device__ __forceinline__ void tdm_load_2d_bf16(unsigned lds_addr, unsigned long long ga,
                                                 int rows, int cols, int stride) {
  v4u g0;
  g0.x = 1u;                                               // count=1, user descriptor
  g0.y = lds_addr;                                         // LDS byte address
  g0.z = (unsigned)(ga & 0xffffffffull);                   // global addr [31:0]
  g0.w = (unsigned)((ga >> 32) & 0x01ffffffull) | 0x80000000u;  // [56:32] | type=2
  v8i g1;
  g1[0] = 0x10000;                                         // data_size=1 (2 bytes)
  g1[1] = (stride & 0xffff) << 16;                         // tensor_dim0 lo16
  g1[2] = ((stride >> 16) & 0xffff) | ((ROWS & 0xffff) << 16); // dim0 hi | dim1 lo
  g1[3] = ((ROWS >> 16) & 0xffff) | (cols << 16);          // dim1 hi | tile_dim0
  g1[4] = rows;                                            // tile_dim1 (tile_dim2=0)
  g1[5] = stride;                                          // tensor_dim0_stride lo32
  g1[6] = 0;
  g1[7] = 0;
  v4i gz = {0, 0, 0, 0};
#if defined(__clang_major__) && (__clang_major__ >= 23)
  v8i gz8 = {0, 0, 0, 0, 0, 0, 0, 0};
  __builtin_amdgcn_tensor_load_to_lds(g0, g1, gz, gz, gz8, 0);
#else
  __builtin_amdgcn_tensor_load_to_lds(g0, g1, gz, gz, 0);
#endif
}
#endif

__device__ __forceinline__ void wait_async_all() {
#if HAVE_ASYNC
#if __has_builtin(__builtin_amdgcn_s_wait_asynccnt)
  __builtin_amdgcn_s_wait_asynccnt(0);
#else
  asm volatile("s_wait_asynccnt 0x0" ::: "memory");
#endif
#endif
}

// ---------------------------------------------------------------------------
// fp32 -> bf16 conversion (weights)
// ---------------------------------------------------------------------------
__global__ void cvt_f32_bf16(const float* __restrict__ in, bf16* __restrict__ out, int n) {
  int i = blockIdx.x * blockDim.x + threadIdx.x;
  if (i < n) out[i] = (bf16)in[i];
}

// ---------------------------------------------------------------------------
// LayerNorm over M=1024, one row per 256-thread block, bf16 output
// ---------------------------------------------------------------------------
__global__ __launch_bounds__(256) void ln_kernel(const float* __restrict__ x,
                                                 const float* __restrict__ sc,
                                                 const float* __restrict__ bi,
                                                 bf16* __restrict__ out) {
  __shared__ float r1[256], r2[256];
  const int row = blockIdx.x, tid = threadIdx.x;
  const float* xp = x + (size_t)row * MDIM;
  float vals[4], s = 0.f, sq = 0.f;
#pragma unroll
  for (int i = 0; i < 4; ++i) {
    float v = xp[tid + 256 * i];
    vals[i] = v; s += v; sq += v * v;
  }
  r1[tid] = s; r2[tid] = sq;
  __syncthreads();
  for (int off = 128; off > 0; off >>= 1) {
    if (tid < off) { r1[tid] += r1[tid + off]; r2[tid] += r2[tid + off]; }
    __syncthreads();
  }
  float mu   = r1[0] * (1.f / MDIM);
  float var  = r2[0] * (1.f / MDIM) - mu * mu;
  float rstd = rsqrtf(var + 1e-5f);
  bf16* op = out + (size_t)row * MDIM;
#pragma unroll
  for (int i = 0; i < 4; ++i) {
    int c = tid + 256 * i;
    op[c] = (bf16)((vals[i] - mu) * rstd * sc[c] + bi[c]);
  }
}

// ---------------------------------------------------------------------------
// Tiled bf16 WMMA GEMM:  out[R x N] = epi( A[R x K] * W[K x N] + bias )
// Block tile 128x128, 8 waves (4 x 2), wave tile 32x64 = 2x4 wmma tiles.
// A tile: async global->LDS b128 copies into padded layout (stride 40 bf16).
// B tile: one TDM tensor_load_to_lds per iteration (raw 32x128), wave 0 issues.
// ---------------------------------------------------------------------------
__global__ __launch_bounds__(256) void gemm_bf16_kernel(
    const bf16*  __restrict__ A, const bf16* __restrict__ W,
    const float* __restrict__ bias, int K, int N,
    float* __restrict__ outF, bf16* __restrict__ outB,
    const float* __restrict__ resid, int dogelu) {
  __shared__ bf16 As[128 * 40];     // padded, 16B-aligned rows (80B stride)
  __shared__ bf16 Bs[32 * 128];     // raw row-major k x n tile

  const int tid = threadIdx.x;
  const int w = tid >> 5, L = tid & 31, half = L >> 4, ln = L & 15;
  const int wm = w & 3, wn = w >> 2;                  // 4 waves along M, 2 along N
  const int rowBase = blockIdx.y * 128, colBase = blockIdx.x * 128;

  v8f acc[2][4];
#pragma unroll
  for (int t = 0; t < 2; ++t)
#pragma unroll
    for (int s = 0; s < 4; ++s) {
      float bv = bias[colBase + wn * 64 + s * 16 + ln];  // per-lane column bias
#pragma unroll
      for (int j = 0; j < 8; ++j) acc[t][s][j] = bv;
    }

  for (int k0 = 0; k0 < K; k0 += 32) {
    // ---- stage A tile 128x32 into padded LDS ----
#if HAVE_ASYNC
#pragma unroll
    for (int c = tid; c < 512; c += 256) {
      int r = c >> 2, seg = c & 3;
      async_copy_b128(A + (size_t)(rowBase + r) * K + k0 + seg * 8,
                      &As[r * 40 + seg * 8]);
    }
#else
#pragma unroll
    for (int c = tid; c < 512; c += 256) {
      int r = c >> 2, seg = c & 3;
      uint4 va = *reinterpret_cast<const uint4*>(A + (size_t)(rowBase + r) * K + k0 + seg * 8);
      *reinterpret_cast<uint4*>(&As[r * 40 + seg * 8]) = va;
    }
#endif

    // ---- stage B tile 32x128 raw ----
#if HAVE_TDM
    if (tid < 32) {   // wave 0 issues the DMA (EXEC-independent, once per wave)
      tdm_load_2d_bf16(lds_offset_of(Bs),
                       (unsigned long long)(size_t)(W + (size_t)k0 * N + colBase),
                       32, 128, N);
    }
#elif HAVE_ASYNC
#pragma unroll
    for (int c = tid; c < 512; c += 256) {
      int kk = c >> 4, seg = c & 15;
      async_copy_b128(W + (size_t)(k0 + kk) * N + colBase + seg * 8,
                      &Bs[kk * 128 + seg * 8]);
    }
#else
#pragma unroll
    for (int c = tid; c < 512; c += 256) {
      int kk = c >> 4, seg = c & 15;
      uint4 vb = *reinterpret_cast<const uint4*>(W + (size_t)(k0 + kk) * N + colBase + seg * 8);
      *reinterpret_cast<uint4*>(&Bs[kk * 128 + seg * 8]) = vb;
    }
#endif

#if HAVE_TDM
    if (tid < 32) __builtin_amdgcn_s_wait_tensorcnt(0);
#endif
    wait_async_all();
    __syncthreads();

    // A fragments: lane=row m, K halves at 8*half and 16+8*half (ISA 16-bit A layout)
    Frag16 af[2], bfr[4];
#pragma unroll
    for (int t = 0; t < 2; ++t) {
      const bf16* p = &As[(wm * 32 + t * 16 + ln) * 40 + 8 * half];
      af[t].u[0] = *reinterpret_cast<const uint4*>(p);
      af[t].u[1] = *reinterpret_cast<const uint4*>(p + 16);
    }
    // B fragments from raw tile: lane=col n, K = 8*half + (e<8 ? e : e+8) (LDS transpose gather)
#pragma unroll
    for (int s = 0; s < 4; ++s) {
      int n = wn * 64 + s * 16 + ln;
#pragma unroll
      for (int e = 0; e < 16; ++e)
        bfr[s].h[e] = Bs[(8 * half + (e < 8 ? e : e + 8)) * 128 + n];
    }
#pragma unroll
    for (int t = 0; t < 2; ++t)
#pragma unroll
      for (int s = 0; s < 4; ++s)
        acc[t][s] = wmma_bf16(af[t].v, bfr[s].v, acc[t][s]);

    if (k0 + 32 < K)
      __builtin_prefetch(W + (size_t)(k0 + 32) * N + colBase + (tid & 31) * 8, 0, 1);
    __syncthreads();
  }

  // epilogue: C/D layout -> row = 16t + j + 8*half, col = 16s + ln
#pragma unroll
  for (int t = 0; t < 2; ++t)
#pragma unroll
    for (int s = 0; s < 4; ++s)
#pragma unroll
      for (int j = 0; j < 8; ++j) {
        int grow = rowBase + wm * 32 + t * 16 + j + 8 * half;
        int gcol = colBase + wn * 64 + s * 16 + ln;
        size_t o = (size_t)grow * N + gcol;
        float v = acc[t][s][j];
        if (resid)  v += resid[o];
        if (dogelu) v = gelu_tanh(v);
        if (outF) outF[o] = v;
        if (outB) outB[o] = (bf16)v;
      }
}

// ---------------------------------------------------------------------------
// Causal attention, one (b, h, 16-query tile) per 128-thread block (4 waves).
// Phase 1: scores = (Q K^T)/8 via 2 chained WMMAs (D=64), masked, into LDS.
// Phase 2: fp32 softmax per row.  Phase 3: out = P V via WMMA, bf16 store.
// ---------------------------------------------------------------------------
__global__ __launch_bounds__(128) void attn_kernel(const bf16* __restrict__ qkv,
                                                   bf16* __restrict__ attn_out) {
  __shared__ float sc[16 * SEQ];  // 64 KB
  const int b = blockIdx.z, h = blockIdx.y, q0 = blockIdx.x * 16;
  const int tid = threadIdx.x, w = tid >> 5, L = tid & 31, half = L >> 4, ln = L & 15;

  const bf16* qb = qkv + (size_t)(b * SEQ) * LDQKV + h * HD;
  const bf16* kb = qb + MDIM;
  const bf16* vb = qb + 2 * MDIM;

  // Q fragments (A layout): lane = query row, d split per ISA 16-bit A layout
  Frag16 qA[2];
  {
    const bf16* qp = qb + (size_t)(q0 + ln) * LDQKV;
#pragma unroll
    for (int f = 0; f < 2; ++f) {
      const bf16* p = qp + 32 * f + 8 * half;
      qA[f].u[0] = *reinterpret_cast<const uint4*>(p);
      qA[f].u[1] = *reinterpret_cast<const uint4*>(p + 16);
    }
  }

  const int klen = q0 + 16;          // causal: keys [0, klen)
  const int ntile = klen >> 4;

  // ---- phase 1: score tiles, 4-way key split across waves ----
  for (int kt = w; kt < ntile; kt += 4) {
    const int key0 = kt * 16;
    Frag16 kB[2];
    const bf16* kp = kb + (size_t)(key0 + ln) * LDQKV;  // lane = key column n
#pragma unroll
    for (int f = 0; f < 2; ++f) {
      kB[f].u[0] = *reinterpret_cast<const uint4*>(kp + 32 * f + 8 * half);
      kB[f].u[1] = *reinterpret_cast<const uint4*>(kp + 32 * f + 8 * half + 16);
    }
    v8f c = {0.f, 0.f, 0.f, 0.f, 0.f, 0.f, 0.f, 0.f};
    c = wmma_bf16(qA[0].v, kB[0].v, c);
    c = wmma_bf16(qA[1].v, kB[1].v, c);
#pragma unroll
    for (int j = 0; j < 8; ++j) {
      int m = j + 8 * half, key = key0 + ln;
      float sv = c[j] * 0.125f;               // 1/sqrt(64)
      if (key > q0 + m) sv = -1e30f;          // causal mask
      sc[m * SEQ + key] = sv;
    }
  }
  __syncthreads();

  // ---- phase 2: fp32 softmax, one thread per query row ----
  if (tid < 16) {
    float* r = &sc[tid * SEQ];
    float mx = -1e30f;
    for (int i = 0; i < klen; ++i) mx = fmaxf(mx, r[i]);
    float sum = 0.f;
    for (int i = 0; i < klen; ++i) { float e = __expf(r[i] - mx); r[i] = e; sum += e; }
    float inv = 1.0f / sum;
    for (int i = 0; i < klen; ++i) r[i] *= inv;
  }
  __syncthreads();

  // ---- phase 3: out(16x64) = P(16xklen) * V(klenx64), wave w owns d cols [16w,16w+16)
  v8f acc = {0.f, 0.f, 0.f, 0.f, 0.f, 0.f, 0.f, 0.f};
  const int steps = (klen + 31) >> 5;
  for (int kk = 0; kk < steps; ++kk) {
    Frag16 pa, vf;
#pragma unroll
    for (int e = 0; e < 16; ++e) {                 // P: A layout, lane = row m = ln
      int key = 32 * kk + 8 * half + (e < 8 ? e : e + 8);
      float pv = (key < klen) ? sc[ln * SEQ + key] : 0.f;
      pa.h[e] = (bf16)pv;
    }
#pragma unroll
    for (int e = 0; e < 16; ++e) {                 // V: B layout, lane = col d
      int key = 32 * kk + 8 * half + (e < 8 ? e : e + 8);
      vf.h[e] = (key < klen) ? vb[(size_t)key * LDQKV + 16 * w + ln] : (bf16)0.f;
    }
    acc = wmma_bf16(pa.v, vf.v, acc);
  }
#pragma unroll
  for (int j = 0; j < 8; ++j) {
    int m = j + 8 * half, d = 16 * w + ln;
    attn_out[(size_t)(b * SEQ + q0 + m) * MDIM + h * HD + d] = (bf16)acc[j];
  }
}

// ---------------------------------------------------------------------------
// Host: orchestrate the 7-stage pipeline on `stream`
// ---------------------------------------------------------------------------
extern "C" void kernel_launch(void* const* d_in, const int* in_sizes, int n_in,
                              void* d_out, int out_size, void* d_ws, size_t ws_size,
                              hipStream_t stream) {
  (void)in_sizes; (void)n_in; (void)out_size; (void)ws_size;
  const float* x     = (const float*)d_in[0];
  const float* w_qkv = (const float*)d_in[1];
  const float* b_qkv = (const float*)d_in[2];
  const float* w_o   = (const float*)d_in[3];
  const float* b_o   = (const float*)d_in[4];
  const float* ln1_s = (const float*)d_in[5];
  const float* ln1_b = (const float*)d_in[6];
  const float* ln2_s = (const float*)d_in[7];
  const float* ln2_b = (const float*)d_in[8];
  const float* w_fc  = (const float*)d_in[9];
  const float* b_fc  = (const float*)d_in[10];
  const float* w_pr  = (const float*)d_in[11];
  const float* b_pr  = (const float*)d_in[12];
  float* out = (float*)d_out;
  char*  ws  = (char*)d_ws;

  // workspace layout (bytes); h_bf reuses qkv+attn region (dead by then). 64 MB total.
  bf16* wqkv_bf = (bf16*)(ws + 0);         // 3072*1024*2 = 6291456
  bf16* wo_bf   = (bf16*)(ws + 6291456);   // 1024*1024*2 = 2097152
  bf16* wfc_bf  = (bf16*)(ws + 8388608);   // 1024*4096*2 = 8388608
  bf16* wpr_bf  = (bf16*)(ws + 16777216);  // 4096*1024*2 = 8388608
  bf16* xn_bf   = (bf16*)(ws + 25165824);  // 4096*1024*2 = 8388608
  bf16* qkv_bf  = (bf16*)(ws + 33554432);  // 4096*3072*2 = 25165824
  bf16* attn_bf = (bf16*)(ws + 58720256);  // 4096*1024*2 = 8388608
  bf16* h_bf    = (bf16*)(ws + 33554432);  // 4096*4096*2 = 33554432 (reuse)

  cvt_f32_bf16<<<12288, 256, 0, stream>>>(w_qkv, wqkv_bf, 3145728);
  cvt_f32_bf16<<< 4096, 256, 0, stream>>>(w_o,   wo_bf,   1048576);
  cvt_f32_bf16<<<16384, 256, 0, stream>>>(w_fc,  wfc_bf,  4194304);
  cvt_f32_bf16<<<16384, 256, 0, stream>>>(w_pr,  wpr_bf,  4194304);

  // ln1 -> qkv GEMM -> attention -> o-proj (+x residual into d_out)
  ln_kernel<<<ROWS, 256, 0, stream>>>(x, ln1_s, ln1_b, xn_bf);
  gemm_bf16_kernel<<<dim3(24, 32), 256, 0, stream>>>(xn_bf, wqkv_bf, b_qkv,
      MDIM, LDQKV, nullptr, qkv_bf, nullptr, 0);
  attn_kernel<<<dim3(SEQ / 16, NH, NB), 128, 0, stream>>>(qkv_bf, attn_bf);
  gemm_bf16_kernel<<<dim3(8, 32), 256, 0, stream>>>(attn_bf, wo_bf, b_o,
      MDIM, MDIM, out, nullptr, x, 0);

  // ln2 -> FC (+GELU) -> proj (+x1 residual into d_out)
  ln_kernel<<<ROWS, 256, 0, stream>>>(out, ln2_s, ln2_b, xn_bf);
  gemm_bf16_kernel<<<dim3(32, 32), 256, 0, stream>>>(xn_bf, wfc_bf, b_fc,
      MDIM, FDIM, nullptr, h_bf, nullptr, 1);
  gemm_bf16_kernel<<<dim3(8, 32), 256, 0, stream>>>(h_bf, wpr_bf, b_pr,
      FDIM, MDIM, out, nullptr, out, 0);
}
